// SimCSELoss_63874753626115
// MI455X (gfx1250) — matrix operand order repaired and checked by reading, not compiled, not run on hardware
//
#include <hip/hip_runtime.h>
#include <hip/hip_bf16.h>
#include <cmath>
#include <cstdint>

typedef _Float16 v16h __attribute__((ext_vector_type(16)));
typedef _Float16 v8h  __attribute__((ext_vector_type(8)));
typedef float    v8f  __attribute__((ext_vector_type(8)));

#define B_ROWS 4096
#define N2     8192
#define DDIM   1024
#define INV_T  10.0f

#define MT 128          // rows per workgroup
#define NT 128          // cols per inner tile
#define KC 64           // K chunk staged in LDS
#define LDS_STRIDE 80   // padded row stride (halves): 160B rows, 32B aligned

#if defined(__has_builtin)
#if __has_builtin(__builtin_amdgcn_tensor_load_to_lds) && __has_builtin(__builtin_amdgcn_s_wait_tensorcnt)
#define HAVE_TDM 1
#endif
#endif
#ifndef HAVE_TDM
#define HAVE_TDM 0
#endif

// Escape hatch: the TDM writes LDS behind the compiler's back (the LDS address
// is descriptor *data*, not a pointer operand), so without this LLVM proves the
// shared array is never stored and folds all ds_loads from it to undef.
// Passing the pointer into opaque asm with a memory clobber forces the
// compiler to treat the LDS contents as externally modified. Emits no code.
__device__ __forceinline__ void lds_escape(const void* p) {
    asm volatile("" : : "r"(p) : "memory");
}

// ---------------------------------------------------------------------------
// Kernel 1: L2-normalize both projections, emit f16 z[8192][1024],
// positives[r] = dot(z_i[r], z_j[r]), and accumulate sum ||z_i - z_j||^2.
// ---------------------------------------------------------------------------
__global__ __launch_bounds__(256) void normalize_kernel(
    const float* __restrict__ p1, const float* __restrict__ p2,
    _Float16* __restrict__ zh, float* __restrict__ pos,
    float* __restrict__ lalign_acc)
{
    const int r = blockIdx.x;
    const int t = threadIdx.x;
    const int wid = t >> 5, lane = t & 31;
    __shared__ float red[16];

    float x1[4], x2[4];
    float ss1 = 0.f, ss2 = 0.f;
#pragma unroll
    for (int q = 0; q < 4; ++q) {
        const int idx = t + q * 256;
        x1[q] = p1[(size_t)r * DDIM + idx];
        x2[q] = p2[(size_t)r * DDIM + idx];
        ss1 += x1[q] * x1[q];
        ss2 += x2[q] * x2[q];
    }
#pragma unroll
    for (int m = 16; m >= 1; m >>= 1) {
        ss1 += __shfl_xor(ss1, m, 32);
        ss2 += __shfl_xor(ss2, m, 32);
    }
    if (lane == 0) { red[wid] = ss1; red[wid + 8] = ss2; }
    __syncthreads();
    if (t == 0) {
        float a = 0.f, b = 0.f;
        for (int w = 0; w < 8; ++w) { a += red[w]; b += red[w + 8]; }
        red[0] = a; red[8] = b;
    }
    __syncthreads();
    const float inv1 = 1.0f / fmaxf(sqrtf(red[0]), 1e-12f);
    const float inv2 = 1.0f / fmaxf(sqrtf(red[8]), 1e-12f);

    float dotp = 0.f, dif = 0.f;
#pragma unroll
    for (int q = 0; q < 4; ++q) {
        const int idx = t + q * 256;
        const float a = x1[q] * inv1;
        const float c = x2[q] * inv2;
        zh[(size_t)r * DDIM + idx]            = (_Float16)a;
        zh[(size_t)(r + B_ROWS) * DDIM + idx] = (_Float16)c;
        dotp += a * c;
        const float dd = a - c;
        dif += dd * dd;
    }
#pragma unroll
    for (int m = 16; m >= 1; m >>= 1) {
        dotp += __shfl_xor(dotp, m, 32);
        dif  += __shfl_xor(dif, m, 32);
    }
    __syncthreads();   // all reads of red[] done before reuse
    if (lane == 0) { red[wid] = dotp; red[wid + 8] = dif; }
    __syncthreads();
    if (t == 0) {
        float a = 0.f, b = 0.f;
        for (int w = 0; w < 8; ++w) { a += red[w]; b += red[w + 8]; }
        pos[r] = a;
        atomicAdd(lalign_acc, b);
    }
}

// ---------------------------------------------------------------------------
// TDM: issue a 2D tensor_load_to_lds of a 128-row x 64-half tile
// (row stride 1024 halves in memory) into LDS at lds_byte_off, with
// hardware LDS padding of 16 halves per row -> effective LDS stride 80 halves.
// Descriptor layout per CDNA5 ISA.
// This toolchain exposes the 6-arg builtin form:
//   (uint32x4 g0, int32x8 g1, int32x4 g2, int32x4 g3, int32x8 extra, i32 cpol)
// groups 2/3 and the extra group are zero for a plain 2-D tensor.
// ---------------------------------------------------------------------------
#if HAVE_TDM
typedef unsigned int tdm_v4u __attribute__((ext_vector_type(4)));
typedef int          tdm_v8i __attribute__((ext_vector_type(8)));
typedef int          tdm_v4i __attribute__((ext_vector_type(4)));

__device__ __forceinline__ void tdm_load_tile_128x64(
    const _Float16* gsrc, unsigned lds_byte_off)
{
    const unsigned long long ga = (unsigned long long)(uintptr_t)gsrc;
    tdm_v4u g0;
    g0[0] = 1u;                                   // count=1 (valid), no gather
    g0[1] = lds_byte_off;                         // lds_addr (bytes)
    g0[2] = (unsigned)ga;                         // global_addr[31:0]
    g0[3] = (unsigned)((ga >> 32) & 0x01FFFFFFu)  // global_addr[56:32]
          | (2u << 30);                           // type = 2 ("image")
    tdm_v8i g1;
    g1[0] = (int)((1u << 16)      // data_size = 1 -> 2 bytes/elem
          | (1u << 20)            // pad_enable
          | (4u << 22)            // pad_interval: 32 DWORDs (one 64-half row)
          | (7u << 25));          // pad_amount: 8 DWORDs (16 halves)
    g1[1] = (int)(((unsigned)DDIM & 0xFFFFu) << 16);          // tensor_dim0 lo
    g1[2] = (int)((((unsigned)DDIM >> 16) & 0xFFFFu)          // tensor_dim0 hi
          | (((unsigned)N2 & 0xFFFFu) << 16));                // tensor_dim1 lo
    g1[3] = (int)((((unsigned)N2 >> 16) & 0xFFFFu)            // tensor_dim1 hi
          | ((unsigned)KC << 16));                            // tile_dim0 = 64
    g1[4] = (int)(MT);             // tile_dim1 = 128, tile_dim2 = 0
    g1[5] = (int)(DDIM);           // tensor_dim0_stride[31:0] = 1024
    g1[6] = 0;                     // stride0 hi | tensor_dim1_stride lo
    g1[7] = 0;
    const tdm_v4i gz4 = {0, 0, 0, 0};            // groups 2/3 unused (2D)
    const tdm_v8i gz8 = {0, 0, 0, 0, 0, 0, 0, 0};
    __builtin_amdgcn_tensor_load_to_lds(g0, g1, gz4, gz4, gz8, 0);
}
#endif

// ---------------------------------------------------------------------------
// Kernel 2: fused GEMM-reduction. For A rows [Mbase..Mbase+127] and B "cols"
// (= rows of Brows) [Cbase..Cbase+colsPerChunk):
//   rowsum[row] += sum_cols exp(scale * dot(Arow, Bcol))
// sim tile never leaves registers. wave32, v_wmma_f32_16x16x32_f16.
// TDM path: wave 0 drives a double-buffered tensor DMA pipeline.
// ---------------------------------------------------------------------------
__global__ __launch_bounds__(256) void rowsum_exp_wmma(
    const _Float16* __restrict__ Arows, const _Float16* __restrict__ Brows,
    int colsPerChunk, float scale, float* __restrict__ rowsum)
{
#if HAVE_TDM
    __shared__ _Float16 Sh[2][2][MT][LDS_STRIDE];   // [buf][A/B][row][k], 80KB
#else
    __shared__ _Float16 Sh[1][2][MT][LDS_STRIDE];
#endif

    const int t    = threadIdx.x;
    const int wave = t >> 5;
    const int lane = t & 31;
    const int mrow = wave * 16;      // wave's 16-row strip inside tile
    const int mA   = lane & 15;
    const int hi   = lane >> 4;

    const int Mbase = blockIdx.x * MT;
    const int Cbase = blockIdx.y * colsPerChunk;
    const int nTiles = colsPerChunk / NT;

    float rs[8];
#pragma unroll
    for (int v = 0; v < 8; ++v) rs[v] = 0.f;

#if HAVE_TDM
    const unsigned tileBytes = (unsigned)(MT * LDS_STRIDE * sizeof(_Float16)); // 20480
    const int totalChunks = nTiles * (DDIM / KC);
    int cur = 0;
    int s = 0;
    if (wave == 0) {   // prologue: chunk 0 -> buffer 0
        tdm_load_tile_128x64(Arows + (size_t)Mbase * DDIM, 0u);
        tdm_load_tile_128x64(Brows + (size_t)Cbase * DDIM, tileBytes);
    }
#else
    const int ldRow = t >> 1;        // 0..127 (row this thread stages)
    const int ldSeg = (t & 1) * 32;  // 0 or 32 (32-half segment)
#endif

    for (int ct = 0; ct < nTiles; ++ct) {
        const int Nbase = Cbase + ct * NT;
        (void)Nbase;

        v8f c[8];
#pragma unroll
        for (int n = 0; n < 8; ++n) c[n] = v8f{};

        for (int kc = 0; kc < DDIM; kc += KC) {
#if HAVE_TDM
            if (wave == 0)
                __builtin_amdgcn_s_wait_tensorcnt(0);   // buffer `cur` filled
            __syncthreads();   // everyone past previous compute; `cur` visible
            lds_escape(&Sh[0][0][0][0]);   // TDM wrote LDS: forbid load folding
            if (wave == 0 && (s + 1) < totalChunks) {
                const int sn  = s + 1;
                const int ctn = sn >> 4;               // 16 K-chunks per tile
                const int kcn = (sn & 15) * KC;
                const unsigned boff = (unsigned)(cur ^ 1) * 2u * tileBytes;
                tdm_load_tile_128x64(Arows + (size_t)Mbase * DDIM + kcn, boff);
                tdm_load_tile_128x64(
                    Brows + (size_t)(Cbase + ctn * NT) * DDIM + kcn,
                    boff + tileBytes);
            }
            const int bsel = cur;
#else
            // cooperative stage: 128 rows x 64 halves each for A and B tiles
            {
                const _Float16* ga = Arows + (size_t)(Mbase + ldRow) * DDIM + kc + ldSeg;
                const _Float16* gb = Brows + (size_t)(Nbase + ldRow) * DDIM + kc + ldSeg;
#pragma unroll
                for (int q = 0; q < 4; ++q) {
                    *(v8h*)&Sh[0][0][ldRow][ldSeg + q * 8] = *(const v8h*)(ga + q * 8);
                    *(v8h*)&Sh[0][1][ldRow][ldSeg + q * 8] = *(const v8h*)(gb + q * 8);
                }
            }
            __syncthreads();
            const int bsel = 0;
#endif
            const _Float16 (*As)[LDS_STRIDE] = Sh[bsel][0];
            const _Float16 (*Bs)[LDS_STRIDE] = Sh[bsel][1];

#pragma unroll
            for (int ks = 0; ks < 2; ++ks) {
                const int k0 = ks * 32;
                // A fragment (16x32 f16): lane m holds K {hi*16..+15}
                v16h a;
                {
                    a = *(const v16h*)&As[mrow + mA][k0 + hi * 16];
                }
                // load all 8 B fragments first so ds_load latency overlaps
                // the WMMA chain (partial dscnt waits instead of full stalls)
                v16h bf[8];
#pragma unroll
                for (int n = 0; n < 8; ++n)
                    bf[n] = *(const v16h*)&Bs[n * 16 + mA][k0 + hi * 16];
#pragma unroll
                for (int n = 0; n < 8; ++n)
                    c[n] = __builtin_amdgcn_wmma_f32_16x16x32_f16(
                        false, a, false, bf[n], (short)0, c[n], false, false);
            }
#if HAVE_TDM
            cur ^= 1; ++s;
            // no trailing barrier: next iteration's top barrier protects buffers
#else
            __syncthreads();
#endif
        }

        // fused epilogue: exp + per-row accumulation (row = v + 8*hi of strip)
#pragma unroll
        for (int n = 0; n < 8; ++n)
#pragma unroll
            for (int v = 0; v < 8; ++v)
                rs[v] += __expf(scale * c[n][v]);
    }

    // reduce across the 16 lanes holding different columns of the same row
#pragma unroll
    for (int v = 0; v < 8; ++v) {
        float x = rs[v];
        x += __shfl_xor(x, 1, 32);
        x += __shfl_xor(x, 2, 32);
        x += __shfl_xor(x, 4, 32);
        x += __shfl_xor(x, 8, 32);
        rs[v] = x;
    }
    if ((lane & 15) == 0) {
        const int rbase = Mbase + mrow + hi * 8;
#pragma unroll
        for (int v = 0; v < 8; ++v)
            atomicAdd(&rowsum[rbase + v], rs[v]);
    }
}

// ---------------------------------------------------------------------------
// Kernel 3: fold everything into the 3 output scalars.
// ---------------------------------------------------------------------------
__device__ __forceinline__ float block_sum256(float x, float* red) {
#pragma unroll
    for (int m = 16; m >= 1; m >>= 1) x += __shfl_xor(x, m, 32);
    const int wid = threadIdx.x >> 5, lane = threadIdx.x & 31;
    __syncthreads();
    if (lane == 0) red[wid] = x;
    __syncthreads();
    float s = 0.f;
    for (int w = 0; w < 8; ++w) s += red[w];
    return s;
}

__global__ __launch_bounds__(256) void finalize_kernel(
    const float* __restrict__ denom_raw, const float* __restrict__ pos,
    const float* __restrict__ unif_rows, const float* __restrict__ lalign_acc,
    float* __restrict__ out)
{
    __shared__ float red[8];
    const int t = threadIdx.x;
    const float EXPD = __expf(INV_T);   // diagonal term exp(sim_ii / T), sim_ii = 1

    float s_logden = 0.f;
    for (int i = t; i < N2; i += 256) {
        const float d = denom_raw[i] - EXPD;
        s_logden += __logf(fmaxf(d, 1e-30f));
    }
    float s_pos = 0.f;
    for (int i = t; i < B_ROWS; i += 256) s_pos += pos[i];
    float u1 = 0.f, u2 = 0.f;
    for (int i = t; i < 2048; i += 256) {
        u1 += unif_rows[i];
        u2 += unif_rows[i + 2048];
    }

    const float S_logden = block_sum256(s_logden, red);
    const float S_pos    = block_sum256(s_pos, red);
    const float U1       = block_sum256(u1, red);
    const float U2       = block_sum256(u2, red);

    if (t == 0) {
        // loss: positives appear twice (sim_ij == sim_ji)
        out[0] = (S_logden - 2.0f * INV_T * S_pos) / (float)N2;
        out[1] = lalign_acc[0] / (float)B_ROWS;
        const float n     = 2048.0f;
        const float pairs = n * (n - 1.0f) * 0.5f;
        const float S1 = fmaxf(0.5f * (U1 * __expf(-4.0f) - n), 1e-30f);
        const float S2 = fmaxf(0.5f * (U2 * __expf(-4.0f) - n), 1e-30f);
        out[2] = 0.5f * (__logf(S1 / pairs) + __logf(S2 / pairs));
    }
}

// ---------------------------------------------------------------------------
extern "C" void kernel_launch(void* const* d_in, const int* in_sizes, int n_in,
                              void* d_out, int out_size, void* d_ws, size_t ws_size,
                              hipStream_t stream) {
    (void)in_sizes; (void)n_in; (void)out_size; (void)ws_size;
    const float* p1 = (const float*)d_in[0];
    const float* p2 = (const float*)d_in[1];
    float* out = (float*)d_out;

    char* ws = (char*)d_ws;
    _Float16* zh = (_Float16*)ws;                       // 8192*1024*2 = 16 MB
    size_t off = (size_t)N2 * DDIM * sizeof(_Float16);
    float* pos        = (float*)(ws + off); off += (size_t)B_ROWS * sizeof(float);
    float* denom_raw  = (float*)(ws + off); off += (size_t)N2 * sizeof(float);
    float* unif_rows  = (float*)(ws + off); off += (size_t)4096 * sizeof(float);
    float* lalign_acc = (float*)(ws + off); off += sizeof(float);

    // zero the atomic accumulators (denom_raw, unif_rows, lalign contiguous)
    (void)hipMemsetAsync(denom_raw, 0,
                         (size_t)(N2 + 4096 + 1) * sizeof(float), stream);

    normalize_kernel<<<B_ROWS, 256, 0, stream>>>(p1, p2, zh, pos, lalign_acc);

    // main softmax denominator: 8192x8192 sim, scale = 1/T = 10
    rowsum_exp_wmma<<<dim3(N2 / MT, 8), 256, 0, stream>>>(
        zh, zh, 1024, INV_T, denom_raw);

    // lunif grams: z_i[0:2048] and z_i[2048:4096], scale = 4 (exp(4g-4) split)
    rowsum_exp_wmma<<<dim3(16, 2), 256, 0, stream>>>(
        zh, zh, 1024, 4.0f, unif_rows);
    rowsum_exp_wmma<<<dim3(16, 2), 256, 0, stream>>>(
        zh + (size_t)2048 * DDIM, zh + (size_t)2048 * DDIM, 1024, 4.0f,
        unif_rows + 2048);

    finalize_kernel<<<1, 256, 0, stream>>>(denom_raw, pos, unif_rows, lalign_acc, out);
}
